// Net_2078764172140
// MI455X (gfx1250) — matrix operand order repaired and checked by reading, not compiled, not run on hardware
//
#include <hip/hip_runtime.h>

// ---------------------------------------------------------------------------
// Types for CDNA5 WMMA (wave32)
// ---------------------------------------------------------------------------
typedef __attribute__((ext_vector_type(16))) __bf16 bf16x16;
typedef __attribute__((ext_vector_type(8)))  __bf16 bf16x8;
typedef __attribute__((ext_vector_type(8)))  float  f32x8;

__device__ __forceinline__ __bf16 f2bf(float x) {
  union { float f; unsigned u; } v; v.f = x;
  unsigned r = v.u + 0x7FFFu + ((v.u >> 16) & 1u);   // round-to-nearest-even
  unsigned short h = (unsigned short)(r >> 16);
  __bf16 o; __builtin_memcpy(&o, &h, 2); return o;
}

// ---------------------------------------------------------------------------
// bf16 WMMA GEMM:  C[M,Np](f32) = A[M,K](bf16) * Bt[Np,K](bf16)^T
// Register-blocked: each wave computes 32(M) x 64(N) = 2 x 4 tiles of 16x16.
// Software-pipelined one K-step deep: fragments for step k are loaded during
// step k-32, so WMMAs never wait on just-issued loads (partial loadcnt waits).
// Bt stored transposed ([N,K] row-major): lane's B fragment = one 32B load.
// A fragment: lane (m = l&15, half = l>>4) loads K
// chunks [16*half, +8) and [16*half+8, +8) (contiguous, matches B).
// epi: 0 = none, 1 = +bias, 2 = relu((x+bias)*g+beta). Cols >= n_real -> 0.
// M, Np multiples of 16; K multiple of 32. Partial tile groups are handled by
// wave-uniform index clamping (loads in-bounds, duplicate stores guarded), so
// EXEC stays all-ones for WMMA.
// ---------------------------------------------------------------------------
__global__ void k_gemm_bf16(const __bf16* __restrict__ A,
                            const __bf16* __restrict__ Bt,
                            float* __restrict__ C,
                            int M, int Np, int K,
                            int lda, int ldb, int ldc, int n_real,
                            const float* __restrict__ bias,
                            const float* __restrict__ gg,
                            const float* __restrict__ beta, int epi) {
  const int lane   = threadIdx.x & 31;
  const int wv     = threadIdx.x >> 5;         // 4 waves / block
  const int ntiles = Np >> 4;
  const int mtiles = M >> 4;
  const int tn0    = (blockIdx.x * 4 + wv) * 4;
  if (tn0 >= ntiles) return;                   // wave-uniform exit
  const int tm0 = blockIdx.y * 2;
  const int mr  = lane & 15;
  const int hf  = lane >> 4;

  int tn[4], tm[2];
#pragma unroll
  for (int t = 0; t < 4; ++t) { int v = tn0 + t; tn[t] = (v < ntiles) ? v : ntiles - 1; }
  tm[0] = tm0;
  tm[1] = (tm0 + 1 < mtiles) ? tm0 + 1 : mtiles - 1;

  const __bf16* ar0 = A + (size_t)(tm[0] * 16 + mr) * lda;
  const __bf16* ar1 = A + (size_t)(tm[1] * 16 + mr) * lda;
  const __bf16* br0 = Bt + (size_t)(tn[0] * 16 + mr) * ldb + hf * 16;
  const __bf16* br1 = Bt + (size_t)(tn[1] * 16 + mr) * ldb + hf * 16;
  const __bf16* br2 = Bt + (size_t)(tn[2] * 16 + mr) * ldb + hf * 16;
  const __bf16* br3 = Bt + (size_t)(tn[3] * 16 + mr) * ldb + hf * 16;

  f32x8 acc[8] = {};

  // prologue: load fragments for k0 = 0
  bf16x8  ca0l = *(const bf16x8*)(ar0 + hf * 16);
  bf16x8  ca0h = *(const bf16x8*)(ar0 + 8 + hf * 16);
  bf16x8  ca1l = *(const bf16x8*)(ar1 + hf * 16);
  bf16x8  ca1h = *(const bf16x8*)(ar1 + 8 + hf * 16);
  bf16x16 cb0  = *(const bf16x16*)(br0);
  bf16x16 cb1  = *(const bf16x16*)(br1);
  bf16x16 cb2  = *(const bf16x16*)(br2);
  bf16x16 cb3  = *(const bf16x16*)(br3);

  for (int k0 = 0; k0 < K; k0 += 32) {
    // prefetch next K-step (wrap keeps the final prefetch in-bounds)
    const int kn = (k0 + 32 < K) ? (k0 + 32) : 0;
    bf16x8  na0l = *(const bf16x8*)(ar0 + kn + hf * 16);
    bf16x8  na0h = *(const bf16x8*)(ar0 + kn + 8 + hf * 16);
    bf16x8  na1l = *(const bf16x8*)(ar1 + kn + hf * 16);
    bf16x8  na1h = *(const bf16x8*)(ar1 + kn + 8 + hf * 16);
    bf16x16 nb0  = *(const bf16x16*)(br0 + kn);
    bf16x16 nb1  = *(const bf16x16*)(br1 + kn);
    bf16x16 nb2  = *(const bf16x16*)(br2 + kn);
    bf16x16 nb3  = *(const bf16x16*)(br3 + kn);

    bf16x16 av0, av1;
#pragma unroll
    for (int i = 0; i < 8; ++i) {
      av0[i] = ca0l[i]; av0[8 + i] = ca0h[i];
      av1[i] = ca1l[i]; av1[8 + i] = ca1h[i];
    }
    acc[0] = __builtin_amdgcn_wmma_f32_16x16x32_bf16(false, av0, false, cb0, (short)0, acc[0], false, false);
    acc[4] = __builtin_amdgcn_wmma_f32_16x16x32_bf16(false, av1, false, cb0, (short)0, acc[4], false, false);
    acc[1] = __builtin_amdgcn_wmma_f32_16x16x32_bf16(false, av0, false, cb1, (short)0, acc[1], false, false);
    acc[5] = __builtin_amdgcn_wmma_f32_16x16x32_bf16(false, av1, false, cb1, (short)0, acc[5], false, false);
    acc[2] = __builtin_amdgcn_wmma_f32_16x16x32_bf16(false, av0, false, cb2, (short)0, acc[2], false, false);
    acc[6] = __builtin_amdgcn_wmma_f32_16x16x32_bf16(false, av1, false, cb2, (short)0, acc[6], false, false);
    acc[3] = __builtin_amdgcn_wmma_f32_16x16x32_bf16(false, av0, false, cb3, (short)0, acc[3], false, false);
    acc[7] = __builtin_amdgcn_wmma_f32_16x16x32_bf16(false, av1, false, cb3, (short)0, acc[7], false, false);

    ca0l = na0l; ca0h = na0h; ca1l = na1l; ca1h = na1h;
    cb0 = nb0; cb1 = nb1; cb2 = nb2; cb3 = nb3;
  }

#pragma unroll
  for (int t = 0; t < 4; ++t) {
    if (tn0 + t >= ntiles) break;              // guard duplicate (clamped) tiles
    const int col = tn[t] * 16 + mr;
    const bool valid = (col < n_real);
    float b_ = 0.f, g_ = 1.f, t_ = 0.f;
    if (valid) {
      if (epi >= 1) b_ = bias[col];
      if (epi == 2) { g_ = gg[col]; t_ = beta[col]; }
    }
#pragma unroll
    for (int mi = 0; mi < 2; ++mi) {
      if (tm0 + mi >= mtiles) break;
      f32x8 a = acc[mi * 4 + t];
#pragma unroll
      for (int r = 0; r < 8; ++r) {
        const int row = tm[mi] * 16 + r + 8 * hf;
        float v = a[r];
        if (!valid)        v = 0.f;
        else if (epi == 1) v = v + b_;
        else if (epi == 2) { v = (v + b_) * g_ + t_; v = v > 0.f ? v : 0.f; }
        C[(size_t)row * ldc + col] = v;
      }
    }
  }
}

// ---------------------------------------------------------------------------
// Weight transpose + f32->bf16 with zero padding: W[K,Nout] -> Bt[Np,Kp]
// ---------------------------------------------------------------------------
__global__ void k_wtrans(const float* __restrict__ W, __bf16* __restrict__ Bt,
                         int K, int Nout, int Kp, int Np) {
  int idx = blockIdx.x * blockDim.x + threadIdx.x;
  if (idx >= Np * Kp) return;
  int n = idx / Kp, k = idx - n * Kp;
  float v = (n < Nout && k < K) ? W[(size_t)k * Nout + n] : 0.f;
  Bt[idx] = f2bf(v);
}

// f32 [M,ld_in] (first Kreal cols) -> bf16 [M,Kp] zero-padded
__global__ void k_astage(const float* __restrict__ X, __bf16* __restrict__ Y,
                         int M, int Kreal, int ld_in, int Kp) {
  int idx = blockIdx.x * blockDim.x + threadIdx.x;
  if (idx >= M * Kp) return;
  int m = idx / Kp, k = idx - m * Kp;
  float v = (k < Kreal) ? X[(size_t)m * ld_in + k] : 0.f;
  Y[idx] = f2bf(v);
}

// ---------------------------------------------------------------------------
// delta = relu((rel @ posW + b) * g + beta); rel = [dpos(3), dnrm(3)], K==16
// ---------------------------------------------------------------------------
__global__ void k_delta(const float* __restrict__ pos, const float* __restrict__ nrm,
                        const int* __restrict__ nbr, const float* __restrict__ W6,
                        const float* __restrict__ bb, const float* __restrict__ gg,
                        const float* __restrict__ beta, float* __restrict__ out,
                        int nbase, int nrows, int C) {
  int idx = blockIdx.x * blockDim.x + threadIdx.x;
  if (idx >= nrows * 16 * C) return;
  int c = idx % C; int nk = idx / C;
  int nl = nk >> 4, k = nk & 15;
  int ng = nbase + nl;
  int nb = nbr[ng * 16 + k];
  float d0 = pos[ng * 3 + 0] - pos[nb * 3 + 0];
  float d1 = pos[ng * 3 + 1] - pos[nb * 3 + 1];
  float d2 = pos[ng * 3 + 2] - pos[nb * 3 + 2];
  float d3 = nrm[ng * 3 + 0] - nrm[nb * 3 + 0];
  float d4 = nrm[ng * 3 + 1] - nrm[nb * 3 + 1];
  float d5 = nrm[ng * 3 + 2] - nrm[nb * 3 + 2];
  float a = d0 * W6[c] + d1 * W6[C + c] + d2 * W6[2 * C + c]
          + d3 * W6[3 * C + c] + d4 * W6[4 * C + c] + d5 * W6[5 * C + c];
  a = (a + bb[c]) * gg[c] + beta[c];
  out[idx] = a > 0.f ? a : 0.f;
}

// t = bf16(dst[n] - src[nbr] + delta)  [attn MLP input, chunk-local rows]
__global__ void k_buildt(const float* __restrict__ SDV, int ldS, int offd, int offs,
                         const float* __restrict__ delta, const int* __restrict__ nbr,
                         __bf16* __restrict__ tb, int nbase, int nrows, int C) {
  int idx = blockIdx.x * blockDim.x + threadIdx.x;
  if (idx >= nrows * 16 * C) return;
  int c = idx % C; int nk = idx / C;
  int nl = nk >> 4, k = nk & 15;
  int ng = nbase + nl;
  int nb = nbr[ng * 16 + k];
  float v = SDV[(size_t)ng * ldS + offd + c]
          - SDV[(size_t)nb * ldS + offs + c]
          + delta[idx];
  tb[idx] = f2bf(v);
}

// softmax over K (=16) per (n,c), then sum_k a * (val[nbr]+delta)
__global__ void k_softwsum(const float* __restrict__ alpha, const float* __restrict__ delta,
                           const float* __restrict__ SDV, int ldS, int offv,
                           const int* __restrict__ nbr, float* __restrict__ out,
                           int ldo, int out_off, int nbase, int nrows, int C) {
  int idx = blockIdx.x * blockDim.x + threadIdx.x;
  if (idx >= nrows * C) return;
  int nl = idx / C, c = idx - nl * C;
  int ng = nbase + nl;
  float av[16]; float mx = -3.4e38f;
#pragma unroll
  for (int k = 0; k < 16; ++k) {
    av[k] = alpha[((size_t)(nl * 16 + k)) * C + c];
    mx = fmaxf(mx, av[k]);
  }
  float s = 0.f, acc = 0.f;
#pragma unroll
  for (int k = 0; k < 16; ++k) {
    float e = expf(av[k] - mx); s += e;
    int nb = nbr[ng * 16 + k];
    acc += e * (SDV[(size_t)nb * ldS + offv + c]
              + delta[((size_t)(nl * 16 + k)) * C + c]);
  }
  out[(size_t)ng * ldo + out_off + c] = acc / s;
}

// Row LayerNorm + exact GELU, in place (one block per row, blockDim 256)
__global__ void k_ln_gelu(float* __restrict__ X, const float* __restrict__ g,
                          const float* __restrict__ b, int ld, int C) {
  __shared__ float s1[256], s2[256];
  float* x = X + (size_t)blockIdx.x * ld;
  float a = 0.f, q = 0.f;
  for (int c = threadIdx.x; c < C; c += 256) { float v = x[c]; a += v; q += v * v; }
  s1[threadIdx.x] = a; s2[threadIdx.x] = q; __syncthreads();
  for (int s = 128; s > 0; s >>= 1) {
    if ((int)threadIdx.x < s) {
      s1[threadIdx.x] += s1[threadIdx.x + s];
      s2[threadIdx.x] += s2[threadIdx.x + s];
    }
    __syncthreads();
  }
  float mean = s1[0] / C;
  float var  = s2[0] / C - mean * mean;
  float rstd = rsqrtf(fmaxf(var, 0.f) + 1e-5f);
  for (int c = threadIdx.x; c < C; c += 256) {
    float v = (x[c] - mean) * rstd * g[c] + b[c];
    x[c] = 0.5f * v * (1.f + erff(v * 0.70710678118f));
  }
}

// atom -> residue pooling (atomic accumulate)
__global__ void k_pool_add(const float* __restrict__ H, const float* __restrict__ pos,
                           const int* __restrict__ seg, const int* __restrict__ mask,
                           float* __restrict__ sum, float* __restrict__ psum,
                           float* __restrict__ cnt, int* __restrict__ mres,
                           int Nn, int C) {
  int idx = blockIdx.x * blockDim.x + threadIdx.x;
  if (idx >= Nn * C) return;
  int n = idx / C, c = idx - n * C;
  int s = seg[n];
  atomicAdd(&sum[(size_t)s * C + c], H[idx]);
  if (c == 0) {
    atomicAdd(&cnt[s], 1.f);
    atomicMax(&mres[s], mask[n]);
    atomicAdd(&psum[s * 4 + 0], pos[n * 3 + 0]);
    atomicAdd(&psum[s * 4 + 1], pos[n * 3 + 1]);
    atomicAdd(&psum[s * 4 + 2], pos[n * 3 + 2]);
  }
}

__global__ void k_pool_div(const float* __restrict__ sum, const float* __restrict__ psum,
                           const float* __restrict__ cnt, __bf16* __restrict__ resb,
                           float* __restrict__ aapos, int R, int C) {
  int idx = blockIdx.x * blockDim.x + threadIdx.x;
  if (idx >= R * C) return;
  int r = idx / C, c = idx - r * C;
  float ct = fmaxf(cnt[r], 1.f);
  resb[idx] = f2bf(sum[idx] / ct);
  if (c < 3) aapos[r * 3 + c] = psum[r * 4 + c] / ct;
}

__global__ void k_add_bf16(const float* __restrict__ a, const float* __restrict__ b,
                           __bf16* __restrict__ o, int total) {
  int i = blockIdx.x * blockDim.x + threadIdx.x;
  if (i < total) o[i] = f2bf(a[i] + b[i]);
}

// final head: out = sigmoid(e@Wf + bf + softmax(esm@onlyW+onlyb)[:,1]) * maskf
__global__ void k_final(const float* __restrict__ E3, int ldE,
                        const float* __restrict__ Wf, const float* __restrict__ bf,
                        const float* __restrict__ esm,
                        const float* __restrict__ oW, const float* __restrict__ ob,
                        const int* __restrict__ mres, float* __restrict__ out, int R) {
  int r = blockIdx.x * blockDim.x + threadIdx.x;
  if (r >= R) return;
  float oe = bf[0];
  for (int j = 0; j < 45; ++j) oe += E3[(size_t)r * ldE + j] * Wf[j];
  float l0 = ob[0], l1 = ob[1];
  for (int j = 0; j < 1280; ++j) {
    float v = esm[(size_t)r * 1280 + j];
    l0 += v * oW[j * 2 + 0];
    l1 += v * oW[j * 2 + 1];
  }
  float m = fmaxf(l0, l1);
  float e0 = expf(l0 - m), e1 = expf(l1 - m);
  float p1 = e1 / (e0 + e1);
  float z = oe + p1;
  float sg = 1.f / (1.f + expf(-z));
  out[r] = (mres[r] == 1) ? sg : 0.f;
}

// ---------------------------------------------------------------------------
// Host-side orchestration
// ---------------------------------------------------------------------------
static inline int cdiv(long a, long b) { return (int)((a + b - 1) / b); }

static inline void gemm(hipStream_t s, const __bf16* A, const __bf16* Bt, float* C,
                        int M, int Np, int K, int lda, int ldb, int ldc, int n_real,
                        const float* bias, const float* g, const float* beta, int epi) {
  dim3 grid(cdiv(Np / 16, 16), cdiv(M / 16, 2));   // 4 waves x (2M x 4N tiles)
  k_gemm_bf16<<<grid, 128, 0, s>>>(A, Bt, C, M, Np, K, lda, ldb, ldc,
                                   n_real, bias, g, beta, epi);
}

extern "C" void kernel_launch(void* const* d_in, const int* in_sizes, int n_in,
                              void* d_out, int out_size, void* d_ws, size_t ws_size,
                              hipStream_t stream) {
  (void)in_sizes; (void)n_in; (void)out_size; (void)ws_size;
  const int Na = 20000, R = 2500, Rp = 2512;           // Rp = 157*16
  const int NCH = 5000, RCH = 625;                     // chunking

  // ---- inputs -------------------------------------------------------------
  const float* x0   = (const float*)d_in[0];
  const float* pos  = (const float*)d_in[1];
  const float* nrm  = (const float*)d_in[2];
  const int*   nbrA[3] = {(const int*)d_in[3], (const int*)d_in[4], (const int*)d_in[5]};
  const int*   nbr5 = (const int*)d_in[6];
  const int*   seg  = (const int*)d_in[7];
  const int*   mask = (const int*)d_in[8];
  const float* esm  = (const float*)d_in[9];
  const float* aann = (const float*)d_in[10];
  auto P = [&](int i) -> const float* { return (const float*)d_in[i]; };
  // params: conv blocks of 11 entries (W_src,W_dst,W_lin,pos_W,pos_b,pos_g,
  //         pos_beta,attn_W,attn_b,attn_g,attn_beta)
  const int PB = 11, C5 = PB + 33;                     // conv5 base = 44
  const int NECK_W = 55, NECK_B = 56, NECK_G = 57, NECK_BETA = 58;
  const int ESM_W = 59, ESM_B = 62, ESM_G = 65, ESM_LB = 68;
  const int ESM_WF = 71, ESM_BF = 72, ONLY_W = 73, ONLY_B = 74;

  // ---- workspace layout ---------------------------------------------------
  char* base = (char*)d_ws; size_t cur = 0;
  auto alloc = [&](size_t bytes) -> char* {
    cur = (cur + 255) & ~(size_t)255;
    char* p = base + cur; cur += bytes; return p;
  };
  __bf16* X0b   = (__bf16*)alloc((size_t)Na * 64 * 2);
  __bf16* W9t   = (__bf16*)alloc((size_t)1152 * 64 * 2);
  __bf16* AWt0  = (__bf16*)alloc((size_t)128 * 128 * 2);
  __bf16* AWt1  = (__bf16*)alloc((size_t)128 * 128 * 2);
  __bf16* AWt2  = (__bf16*)alloc((size_t)128 * 128 * 2);
  __bf16* AWt[3] = {AWt0, AWt1, AWt2};
  __bf16* NKt   = (__bf16*)alloc((size_t)1280 * 384 * 2);
  __bf16* W5t   = (__bf16*)alloc((size_t)3840 * 1280 * 2);
  __bf16* A5t   = (__bf16*)alloc((size_t)1280 * 1280 * 2);
  __bf16* E1t   = (__bf16*)alloc((size_t)160 * 1280 * 2);
  __bf16* E2t   = (__bf16*)alloc((size_t)128 * 160 * 2);
  __bf16* E3t   = (__bf16*)alloc((size_t)48 * 128 * 2);
  float*  SDV   = (float*)alloc((size_t)Na * 1152 * 4);    // src|dst|val x3 convs
  float*  xcatF = (float*)alloc((size_t)Na * 384 * 4);
  __bf16* xcatB = (__bf16*)alloc((size_t)Na * 384 * 2);
  float*  rsum  = (float*)alloc((size_t)Rp * 1280 * 4);
  float*  psum  = (float*)alloc((size_t)Rp * 4 * 4);
  float*  cnt   = (float*)alloc((size_t)Rp * 4);
  int*    mres  = (int*)alloc((size_t)Rp * 4);
  __bf16* resb  = (__bf16*)alloc((size_t)Rp * 1280 * 2);
  float*  aapos = (float*)alloc((size_t)Rp * 3 * 4);
  float*  SDV5  = (float*)alloc((size_t)Rp * 3840 * 4);
  float*  res2  = (float*)alloc((size_t)Rp * 1280 * 4);
  __bf16* eb0   = (__bf16*)alloc((size_t)Rp * 1280 * 2);
  float*  Ee1   = (float*)alloc((size_t)Rp * 160 * 4);
  __bf16* eb1   = (__bf16*)alloc((size_t)Rp * 160 * 2);
  float*  Ee2   = (float*)alloc((size_t)Rp * 128 * 4);
  __bf16* eb2   = (__bf16*)alloc((size_t)Rp * 128 * 2);
  float*  Ee3   = (float*)alloc((size_t)Rp * 48 * 4);
  // reusable arena (sized for residue chunks; atom chunks + H overlay fit)
  const size_t szF1 = (size_t)RCH * 16 * 1280 * 4;   // 51.2 MB (delta)
  const size_t szB  = (size_t)RCH * 16 * 1280 * 2;   // 25.6 MB (t bf16)
  float*  arF1 = (float*)alloc(szF1);
  __bf16* arB  = (__bf16*)alloc(szB);
  float*  arF2 = (float*)alloc(szF1);                //        (alpha)
  float*  H    = arF1;                               // 102.4 MB overlay on arena

  const int TB = 256;
  auto G1D = [&](long n) { return dim3(cdiv(n, TB)); };

  // ---- stage W: weight transposes / conversions ---------------------------
  for (int i = 0; i < 3; ++i)
    for (int j = 0; j < 3; ++j)
      k_wtrans<<<G1D(128 * 64), TB, 0, stream>>>(P(PB + 11 * i + j),
          W9t + (size_t)(i * 384 + j * 128) * 64, 59, 128, 64, 128);
  for (int i = 0; i < 3; ++i)
    k_wtrans<<<G1D(128 * 128), TB, 0, stream>>>(P(PB + 11 * i + 7), AWt[i],
                                                128, 128, 128, 128);
  k_wtrans<<<G1D(1280 * 384), TB, 0, stream>>>(P(NECK_W), NKt, 384, 1280, 384, 1280);
  for (int j = 0; j < 3; ++j)
    k_wtrans<<<G1D(1280 * 1280), TB, 0, stream>>>(P(C5 + j),
        W5t + (size_t)j * 1280 * 1280, 1280, 1280, 1280, 1280);
  k_wtrans<<<G1D(1280 * 1280), TB, 0, stream>>>(P(C5 + 7), A5t, 1280, 1280, 1280, 1280);
  k_wtrans<<<G1D(160 * 1280), TB, 0, stream>>>(P(ESM_W + 0), E1t, 1280, 150, 1280, 160);
  k_wtrans<<<G1D(128 * 160), TB, 0, stream>>>(P(ESM_W + 1), E2t, 150, 120, 160, 128);
  k_wtrans<<<G1D(48 * 128), TB, 0, stream>>>(P(ESM_W + 2), E3t, 120, 45, 128, 48);
  k_astage<<<G1D((long)Na * 64), TB, 0, stream>>>(x0, X0b, Na, 59, 59, 64);

  // ---- stage A: src/dst/val for all three atom convs ----------------------
  gemm(stream, X0b, W9t, SDV, Na, 1152, 64, 64, 64, 1152, 1152,
       nullptr, nullptr, nullptr, 0);

  // ---- stage B: three PointTransformerConv branches (chunked) -------------
  for (int i = 0; i < 3; ++i) {
    const int pb = PB + 11 * i;
    for (int ch = 0; ch < Na / NCH; ++ch) {
      const int nb0 = ch * NCH;
      k_delta<<<G1D((long)NCH * 16 * 128), TB, 0, stream>>>(
          pos, nrm, nbrA[i], P(pb + 3), P(pb + 4), P(pb + 5), P(pb + 6),
          arF1, nb0, NCH, 128);
      k_buildt<<<G1D((long)NCH * 16 * 128), TB, 0, stream>>>(
          SDV, 1152, i * 384 + 128, i * 384, arF1, nbrA[i], arB, nb0, NCH, 128);
      gemm(stream, arB, AWt[i], arF2, NCH * 16, 128, 128, 128, 128, 128, 128,
           P(pb + 8), P(pb + 9), P(pb + 10), 2);
      k_softwsum<<<G1D((long)NCH * 128), TB, 0, stream>>>(
          arF2, arF1, SDV, 1152, i * 384 + 256, nbrA[i],
          xcatF, 384, i * 128, nb0, NCH, 128);
    }
  }
  k_astage<<<G1D((long)Na * 384), TB, 0, stream>>>(xcatF, xcatB, Na, 384, 384, 384);

  // ---- stage C: neck linear + LN + GELU -----------------------------------
  gemm(stream, xcatB, NKt, H, Na, 1280, 384, 384, 384, 1280, 1280,
       P(NECK_B), nullptr, nullptr, 1);
  k_ln_gelu<<<Na, TB, 0, stream>>>(H, P(NECK_G), P(NECK_BETA), 1280, 1280);

  // ---- stage D: atom -> residue mean pool + mask max ----------------------
  hipMemsetAsync(rsum, 0, (size_t)Rp * 1280 * 4, stream);
  hipMemsetAsync(psum, 0, (size_t)Rp * 4 * 4, stream);
  hipMemsetAsync(cnt, 0, (size_t)Rp * 4, stream);
  hipMemsetAsync(mres, 0, (size_t)Rp * 4, stream);
  k_pool_add<<<G1D((long)Na * 1280), TB, 0, stream>>>(
      H, pos, seg, mask, rsum, psum, cnt, mres, Na, 1280);
  k_pool_div<<<G1D((long)R * 1280), TB, 0, stream>>>(
      rsum, psum, cnt, resb, aapos, R, 1280);

  // ---- stage E: residue-level PointTransformerConv (conv5) ----------------
  gemm(stream, resb, W5t, SDV5, Rp, 3840, 1280, 1280, 1280, 3840, 3840,
       nullptr, nullptr, nullptr, 0);
  for (int ch = 0; ch < R / RCH; ++ch) {
    const int rb0 = ch * RCH;
    k_delta<<<G1D((long)RCH * 16 * 1280), TB, 0, stream>>>(
        aapos, aann, nbr5, P(C5 + 3), P(C5 + 4), P(C5 + 5), P(C5 + 6),
        arF1, rb0, RCH, 1280);
    k_buildt<<<G1D((long)RCH * 16 * 1280), TB, 0, stream>>>(
        SDV5, 3840, 1280, 0, arF1, nbr5, arB, rb0, RCH, 1280);
    gemm(stream, arB, A5t, arF2, RCH * 16, 1280, 1280, 1280, 1280, 1280, 1280,
         P(C5 + 8), P(C5 + 9), P(C5 + 10), 2);
    k_softwsum<<<G1D((long)RCH * 1280), TB, 0, stream>>>(
        arF2, arF1, SDV5, 3840, 2560, nbr5, res2, 1280, 0, rb0, RCH, 1280);
  }

  // ---- stage F: esm head --------------------------------------------------
  k_add_bf16<<<G1D((long)R * 1280), TB, 0, stream>>>(res2, esm, eb0, R * 1280);
  gemm(stream, eb0, E1t, Ee1, Rp, 160, 1280, 1280, 1280, 160, 150,
       P(ESM_B + 0), nullptr, nullptr, 1);
  k_ln_gelu<<<R, TB, 0, stream>>>(Ee1, P(ESM_G + 0), P(ESM_LB + 0), 160, 150);
  k_astage<<<G1D((long)R * 160), TB, 0, stream>>>(Ee1, eb1, R, 150, 160, 160);
  gemm(stream, eb1, E2t, Ee2, Rp, 128, 160, 160, 160, 128, 120,
       P(ESM_B + 1), nullptr, nullptr, 1);
  k_ln_gelu<<<R, TB, 0, stream>>>(Ee2, P(ESM_G + 1), P(ESM_LB + 1), 128, 120);
  k_astage<<<G1D((long)R * 128), TB, 0, stream>>>(Ee2, eb2, R, 120, 128, 128);
  gemm(stream, eb2, E3t, Ee3, Rp, 48, 128, 128, 128, 48, 45,
       P(ESM_B + 2), nullptr, nullptr, 1);
  k_ln_gelu<<<R, TB, 0, stream>>>(Ee3, P(ESM_G + 2), P(ESM_LB + 2), 48, 45);

  // ---- stage G: final logits ----------------------------------------------
  k_final<<<G1D(R), TB, 0, stream>>>(Ee3, 48, P(ESM_WF), P(ESM_BF), esm,
                                     P(ONLY_W), P(ONLY_B), mres,
                                     (float*)d_out, R);
}
